// GRU_GAT_11364483465461
// MI455X (gfx1250) — matrix operand order repaired and checked by reading, not compiled, not run on hardware
//
#include <hip/hip_runtime.h>
#include <hip/hip_bf16.h>
#include <math.h>

typedef __attribute__((ext_vector_type(16))) _Float16 v16h;
typedef __attribute__((ext_vector_type(8)))  float    v8f;

#define MT 8   // M-subtiles (16 rows each) per wave -> block tile 256 x 64

// ---------------------------------------------------------------------------
// Generic WMMA GEMM:  C[M x N] = gathered-A[M x K] @ B[K x N] (+ bias[n])
// Branch-free inner loop: OOB loads clamped to valid addresses (selects, not
// exec branches); only stores are guarded. Ragged K-tail handled once outside
// the hot loop. A loads vectorized as float4 (global_load_b128); one B
// fragment is amortized over MT=8 WMMAs. Block = 256 threads = 8 waves
// (wave32), waves 2(m) x 4(n); block tile = 256 x 64, so W_g is streamed from
// HBM only ceil(M/256) = 4 times (~820 MB, ~35 us at 23.3 TB/s).
// ---------------------------------------------------------------------------
__global__ void __launch_bounds__(256)
wmma_gemm_f16f32(const float* __restrict__ A, const int* __restrict__ gather,
                 int lda, const float* __restrict__ Bm,
                 const float* __restrict__ bias,
                 float* __restrict__ C, int ldc, int M, int N, int K)
{
    const int lane  = threadIdx.x & 31;     // wave32 on gfx1250
    const int wave  = threadIdx.x >> 5;
    const int wm    = wave >> 2;            // 0..1
    const int wn    = wave & 3;             // 0..3
    const int khalf = lane >> 4;            // 0..1
    const int l16   = lane & 15;

    const int ncol   = blockIdx.x * 64 + wn * 16 + l16;  // this lane's C column
    const int nclamp = (ncol < N) ? ncol : (N - 1);      // always-valid column
    const int mbase  = blockIdx.y * (MT * 32) + wm * (MT * 16);

    // Always-valid row pointers for the MT M-subtiles (fused embedding gather)
    const float* arow[MT];
#pragma unroll
    for (int t = 0; t < MT; ++t) {
        int m  = mbase + t * 16 + l16;
        int mc = (m < M) ? m : (M - 1);
        long r = gather ? (long)gather[mc] : (long)mc;
        arow[t] = A + r * (long)lda;
    }

    v8f acc[MT] = {};
    const int kfull = K & ~31;

    for (int k0 = 0; k0 < kfull; k0 += 32) {
        // Prefetch next K-slab of B (clamped address -> no exec branching)
        long kpre = (long)((k0 + 32 < K) ? (k0 + 32) : k0) + (khalf << 4);
        __builtin_prefetch(Bm + kpre * N + nclamp, 0, 1);

        // ---- B fragment: 32x16 f16. Lane holds column nclamp,
        //      k = k0 + 8*khalf + e + (e & 8) for e in [0,16). Unconditional loads.
        const float* bp = Bm + (long)(k0 + (khalf << 3)) * N + nclamp;
        v16h b;
#pragma unroll
        for (int e = 0; e < 16; ++e)
            b[e] = (_Float16)bp[(long)(e + (e & 8)) * N];

        // ---- A fragments: 16x32 f16, row m = lane%16.
        //      halfs 0..7  -> k0 + 8*khalf + [0,8)   (contiguous, float4 x2)
        //      halfs 8..15 -> k0 + 16 + 8*khalf + [0,8)
#pragma unroll
        for (int t = 0; t < MT; ++t) {
            const float* ap = arow[t] + k0 + (khalf << 3);
            float4 f0 = *(const float4*)(ap);
            float4 f1 = *(const float4*)(ap + 4);
            float4 f2 = *(const float4*)(ap + 16);
            float4 f3 = *(const float4*)(ap + 20);
            v16h a;
            a[0]  = (_Float16)f0.x; a[1]  = (_Float16)f0.y;
            a[2]  = (_Float16)f0.z; a[3]  = (_Float16)f0.w;
            a[4]  = (_Float16)f1.x; a[5]  = (_Float16)f1.y;
            a[6]  = (_Float16)f1.z; a[7]  = (_Float16)f1.w;
            a[8]  = (_Float16)f2.x; a[9]  = (_Float16)f2.y;
            a[10] = (_Float16)f2.z; a[11] = (_Float16)f2.w;
            a[12] = (_Float16)f3.x; a[13] = (_Float16)f3.y;
            a[14] = (_Float16)f3.z; a[15] = (_Float16)f3.w;
            acc[t] = __builtin_amdgcn_wmma_f32_16x16x32_f16(
                         false, a, false, b, (short)0, acc[t], false, false);
        }
    }

    // ---- Ragged K-tail (runs at most once; K=300 path). Clamp addresses,
    //      zero out-of-range values with selects (v_cndmask), no branches.
    if (kfull < K) {
        const int k0 = kfull;
        v16h b;
#pragma unroll
        for (int e = 0; e < 16; ++e) {
            int k  = k0 + (khalf << 3) + e + (e & 8);
            int kc = (k < K) ? k : (K - 1);
            float v = Bm[(long)kc * N + nclamp];
            b[e] = (_Float16)((k < K) ? v : 0.0f);
        }
#pragma unroll
        for (int t = 0; t < MT; ++t) {
            v16h a;
#pragma unroll
            for (int e = 0; e < 8; ++e) {
                int kA = k0 + (khalf << 3) + e;
                int kB = kA + 16;
                int kAc = (kA < K) ? kA : (K - 1);
                int kBc = (kB < K) ? kB : (K - 1);
                float vA = arow[t][kAc];
                float vB = arow[t][kBc];
                a[e]     = (_Float16)((kA < K) ? vA : 0.0f);
                a[e + 8] = (_Float16)((kB < K) ? vB : 0.0f);
            }
            acc[t] = __builtin_amdgcn_wmma_f32_16x16x32_f16(
                         false, a, false, b, (short)0, acc[t], false, false);
        }
    }

    // ---- Store C (guarded only here); C layout: M = r + 8*(lane>=16)
    const float bv = bias ? bias[nclamp] : 0.0f;
#pragma unroll
    for (int t = 0; t < MT; ++t) {
#pragma unroll
        for (int r = 0; r < 8; ++r) {
            int m = mbase + t * 16 + r + (khalf << 3);
            if (m < M && ncol < N)
                C[(long)m * ldc + ncol] = acc[t][r] + bv;
        }
    }
}

// ---------------------------------------------------------------------------
// Sequential 2-layer GRU scan: one workgroup of H=1024 threads, state in LDS.
// Input projections (e@W_*_1) are precomputed in A1z/A1r/A1w by WMMA GEMMs.
// Thread j owns hidden element j; each dot = coalesced column read of U/W.
// ---------------------------------------------------------------------------
#define GRU_H 1024

__device__ __forceinline__ float sigmf(float x) { return 1.0f / (1.0f + __expf(-x)); }

__global__ void __launch_bounds__(GRU_H)
gru_scan(const float* __restrict__ A1z, const float* __restrict__ A1r,
         const float* __restrict__ A1w,
         const float* __restrict__ Uz1, const float* __restrict__ Ur1,
         const float* __restrict__ U1,  const float* __restrict__ bW1,
         const float* __restrict__ bU1,
         const float* __restrict__ Wz2, const float* __restrict__ Uz2,
         const float* __restrict__ Wr2, const float* __restrict__ Ur2,
         const float* __restrict__ W2,  const float* __restrict__ bW2,
         const float* __restrict__ U2,  const float* __restrict__ bU2,
         float* __restrict__ hs, int S)
{
    __shared__ float h1[GRU_H], h2[GRU_H], tmp[GRU_H];
    const int j = threadIdx.x;
    h1[j] = 0.0f; h2[j] = 0.0f;
    __syncthreads();

    for (int s = 0; s < S; ++s) {
        // ---- layer 1, phase 1: z/r gates
        float uz = 0.0f, ur = 0.0f;
#pragma unroll 4
        for (int i = 0; i < GRU_H; ++i) {
            float hv = h1[i];
            uz = fmaf(hv, Uz1[(long)i * GRU_H + j], uz);
            ur = fmaf(hv, Ur1[(long)i * GRU_H + j], ur);
        }
        float z1 = sigmf(A1z[(long)s * GRU_H + j] + uz);
        float r1 = sigmf(A1r[(long)s * GRU_H + j] + ur);
        tmp[j] = r1 * h1[j];
        __syncthreads();

        // ---- layer 1, phase 2: candidate + update
        float u1 = 0.0f;
#pragma unroll 4
        for (int i = 0; i < GRU_H; ++i)
            u1 = fmaf(tmp[i], U1[(long)i * GRU_H + j], u1);
        float ht1 = tanhf(A1w[(long)s * GRU_H + j] + bW1[j] + u1 + bU1[j]);
        float h1n = z1 * ht1 + (1.0f - z1) * h1[j];
        __syncthreads();
        h1[j] = h1n;
        __syncthreads();

        // ---- layer 2, phase 1
        float wz = 0.0f, uz2 = 0.0f, wr = 0.0f, ur2 = 0.0f;
#pragma unroll 2
        for (int i = 0; i < GRU_H; ++i) {
            float a = h1[i], b = h2[i];
            wz  = fmaf(a, Wz2[(long)i * GRU_H + j], wz);
            uz2 = fmaf(b, Uz2[(long)i * GRU_H + j], uz2);
            wr  = fmaf(a, Wr2[(long)i * GRU_H + j], wr);
            ur2 = fmaf(b, Ur2[(long)i * GRU_H + j], ur2);
        }
        float z2 = sigmf(wz + uz2);
        float r2 = sigmf(wr + ur2);
        tmp[j] = r2 * h2[j];
        __syncthreads();

        // ---- layer 2, phase 2
        float w2 = 0.0f, u2 = 0.0f;
#pragma unroll 4
        for (int i = 0; i < GRU_H; ++i) {
            w2 = fmaf(h1[i], W2[(long)i * GRU_H + j], w2);
            u2 = fmaf(tmp[i], U2[(long)i * GRU_H + j], u2);
        }
        float ht2 = tanhf(w2 + bW2[j] + u2 + bU2[j]);
        float h2n = z2 * ht2 + (1.0f - z2) * h2[j];
        __syncthreads();
        h2[j] = h2n;
        hs[(long)s * GRU_H + j] = h2n;
        __syncthreads();
    }
}

// ---------------------------------------------------------------------------
// log-softmax pass 1: per-row stat = max + log(sum(exp(x - max)))
// ---------------------------------------------------------------------------
__global__ void __launch_bounds__(256)
rowstat(const float* __restrict__ logits, float* __restrict__ stat, int N)
{
    const int row = blockIdx.x;
    const float* x = logits + (long)row * N;
    __shared__ float red[256];

    float m = -INFINITY;
    for (int i = threadIdx.x; i < N; i += 256) m = fmaxf(m, x[i]);
    red[threadIdx.x] = m;
    __syncthreads();
    for (int off = 128; off > 0; off >>= 1) {
        if (threadIdx.x < off)
            red[threadIdx.x] = fmaxf(red[threadIdx.x], red[threadIdx.x + off]);
        __syncthreads();
    }
    m = red[0];
    __syncthreads();

    float ssum = 0.0f;
    for (int i = threadIdx.x; i < N; i += 256) ssum += __expf(x[i] - m);
    red[threadIdx.x] = ssum;
    __syncthreads();
    for (int off = 128; off > 0; off >>= 1) {
        if (threadIdx.x < off)
            red[threadIdx.x] += red[threadIdx.x + off];
        __syncthreads();
    }
    if (threadIdx.x == 0) stat[row] = m + logf(red[0]);
}

// log-softmax pass 2: x -= stat[row]
__global__ void __launch_bounds__(256)
sub_stat(float* __restrict__ logits, const float* __restrict__ stat, int N)
{
    const int row = blockIdx.y;
    const int i = blockIdx.x * 256 + threadIdx.x;
    if (i < N) logits[(long)row * N + i] -= stat[row];
}

// ---------------------------------------------------------------------------
extern "C" void kernel_launch(void* const* d_in, const int* in_sizes, int n_in,
                              void* d_out, int out_size, void* d_ws, size_t ws_size,
                              hipStream_t stream)
{
    const int*   x_idx = (const int*)  d_in[0];
    const float* X     = (const float*)d_in[1];
    const float* W_z_1 = (const float*)d_in[2];  const float* U_z_1 = (const float*)d_in[3];
    const float* W_r_1 = (const float*)d_in[4];  const float* U_r_1 = (const float*)d_in[5];
    const float* W_1   = (const float*)d_in[6];  const float* b_W_1 = (const float*)d_in[7];
    const float* U_1   = (const float*)d_in[8];  const float* b_U_1 = (const float*)d_in[9];
    const float* W_z_2 = (const float*)d_in[10]; const float* U_z_2 = (const float*)d_in[11];
    const float* W_r_2 = (const float*)d_in[12]; const float* U_r_2 = (const float*)d_in[13];
    const float* W_2   = (const float*)d_in[14]; const float* b_W_2 = (const float*)d_in[15];
    const float* U_2   = (const float*)d_in[16]; const float* b_U_2 = (const float*)d_in[17];
    const float* W_g   = (const float*)d_in[18]; const float* b_g   = (const float*)d_in[19];
    const float* W_s   = (const float*)d_in[20]; const float* b_s   = (const float*)d_in[21];

    const int S  = in_sizes[0];            // B*T = 1024
    const int H  = in_sizes[7];            // 1024
    const int D  = in_sizes[2] / H;        // 300
    const int NG = in_sizes[19];           // 50000
    const int NS = in_sizes[21];           // 25000

    // Workspace layout (floats): A1z | A1r | A1w | hs | stat_g | stat_s
    float* ws  = (float*)d_ws;
    float* A1z = ws;
    float* A1r = A1z + (size_t)S * H;
    float* A1w = A1r + (size_t)S * H;
    float* hsb = A1w + (size_t)S * H;
    float* stg = hsb + (size_t)S * H;
    float* sts = stg + S;

    float* out_g = (float*)d_out;
    float* out_s = out_g + (size_t)S * NG;

    const dim3 blk(256);
    auto gemm = [&](const float* A, const int* gather, int lda, const float* B,
                    const float* bias, float* C, int ldc, int M, int N, int K) {
        dim3 grid((N + 63) / 64, (M + MT * 32 - 1) / (MT * 32));
        wmma_gemm_f16f32<<<grid, blk, 0, stream>>>(A, gather, lda, B, bias,
                                                   C, ldc, M, N, K);
    };

    // 1) Input projections with fused embedding gather: [S x D] @ [D x H]
    gemm(X, x_idx, D, W_z_1, nullptr, A1z, H, S, H, D);
    gemm(X, x_idx, D, W_r_1, nullptr, A1r, H, S, H, D);
    gemm(X, x_idx, D, W_1,   nullptr, A1w, H, S, H, D);

    // 2) Latency-bound sequential 2-layer GRU scan (single WGP, LDS state)
    gru_scan<<<dim3(1), dim3(H), 0, stream>>>(
        A1z, A1r, A1w, U_z_1, U_r_1, U_1, b_W_1, b_U_1,
        W_z_2, U_z_2, W_r_2, U_r_2, W_2, b_W_2, U_2, b_U_2, hsb, S);

    // 3) Output projections (the 157 GFLOP bulk): hs @ W_g, hs @ W_s (+bias)
    gemm(hsb, nullptr, H, W_g, b_g, out_g, NG, S, NG, H);
    gemm(hsb, nullptr, H, W_s, b_s, out_s, NS, S, NS, H);

    // 4) log-softmax, two passes per matrix
    rowstat<<<dim3(S), dim3(256), 0, stream>>>(out_g, stg, NG);
    rowstat<<<dim3(S), dim3(256), 0, stream>>>(out_s, sts, NS);
    sub_stat<<<dim3((NG + 255) / 256, S), dim3(256), 0, stream>>>(out_g, stg, NG);
    sub_stat<<<dim3((NS + 255) / 256, S), dim3(256), 0, stream>>>(out_s, sts, NS);
}
